// GatedPropagationModel_48533130445171
// MI455X (gfx1250) — matrix operand order repaired and checked by reading, not compiled, not run on hardware
//
#include <hip/hip_runtime.h>
#include <hip/hip_bf16.h>
#include <math.h>

// ---- problem dims (from reference) ----
#define Bn   32
#define Nn   256
#define Kn   32
#define En   1024
#define Dn   256
#define Mtot (Bn * Nn)   // 8192 rows
#define KD   (2 * Dn)    // 512  (concat dim)

#define LDS_STRIDE 136   // 128 K elems + 8 pad (272B rows, 16B aligned)

typedef __attribute__((ext_vector_type(16))) __bf16 v16bf;
typedef __attribute__((ext_vector_type(8)))  __bf16 v8bf;
typedef __attribute__((ext_vector_type(8)))  float  v8f;

static __device__ __forceinline__ __bf16 to_bf16(float f) {
  union { float f; unsigned u; } x; x.f = f;
  unsigned r = x.u + 0x7FFFu + ((x.u >> 16) & 1u);   // round-to-nearest-even
  union { unsigned short s; __bf16 b; } y;
  y.s = (unsigned short)(r >> 16);
  return y.b;
}

static __device__ __forceinline__ float sigmoidf_fast(float x) {
  return 1.0f / (1.0f + __expf(-x));
}

// ---------------------------------------------------------------------------
// Kernel 1: convert weights to transposed bf16.
//   wurt[n][k] = (n<256 ? Wu[k][n] : Wr[k][n-256]),  n in [0,512), k in [0,512)
//   wht [n][k] = Wh[k][n],                           n in [0,256), k in [0,512)
// ---------------------------------------------------------------------------
__global__ __launch_bounds__(256) void prep_weights(
    const float* __restrict__ Wu, const float* __restrict__ Wr,
    const float* __restrict__ Wh,
    __bf16* __restrict__ wurt, __bf16* __restrict__ wht) {
  int tid = blockIdx.x * 256 + threadIdx.x;
  if (tid < 512 * 512) {
    int n = tid >> 9, k = tid & 511;
    float v = (n < Dn) ? Wu[k * Dn + n] : Wr[k * Dn + (n - Dn)];
    wurt[tid] = to_bf16(v);
  } else if (tid < 512 * 512 + 256 * 512) {
    int t = tid - 512 * 512;
    int n = t >> 9, k = t & 511;
    wht[t] = to_bf16(Wh[k * Dn + n]);
  }
}

// ---------------------------------------------------------------------------
// Kernel 2: masked gather-sum over K neighbors, emit bf16 [activation | cs].
// ---------------------------------------------------------------------------
__global__ __launch_bounds__(256) void gather_sum(
    const float* __restrict__ cs, const float* __restrict__ em,
    const int* __restrict__ an, const int* __restrict__ ae,
    __bf16* __restrict__ acsb) {
  const int row = blockIdx.x;        // b*N + n
  const int b   = row >> 8;
  const int d   = threadIdx.x;
  const int*  anp  = an + row * Kn;
  const int*  aep  = ae + row * Kn;
  const float* csb = cs + b * Nn * Dn;
  const float* emb = em + b * En * Dn;
  float acc = 0.0f;
#pragma unroll 4
  for (int k = 0; k < Kn; ++k) {
    int in_ = anp[k];
    int ie  = aep[k];
    if (in_ != 0) acc += csb[in_ * Dn + d];
    if (ie  != 0) acc += emb[ie  * Dn + d];
  }
  acsb[row * KD + d]      = to_bf16(acc);              // activation half
  acsb[row * KD + Dn + d] = to_bf16(cs[row * Dn + d]); // current_state half
}

// ---- WMMA fragment helpers (contiguous K order used by this file, wave32) ------------------
// A (16x32 bf16): lane l -> row l&15; h=l>>4; elem i -> K = i + 16h
//   => two contiguous 16B loads at base+16h and base+16h+8.
// B (32x16 bf16): lane l -> col l&15;  elem i -> K = i + 16h
//   => two contiguous 16B loads at base (LDS, transposed weights).
union FragU { v16bf v; v8bf p[2]; };

static __device__ __forceinline__ v16bf load_a_frag(const __bf16* __restrict__ base,
                                                    int h) {
  FragU a;
  a.p[0] = *(const v8bf*)(base + 16 * h);
  a.p[1] = *(const v8bf*)(base + 16 * h + 8);
  return a.v;
}
static __device__ __forceinline__ v16bf load_b_lds(const __bf16* p) {
  FragU b;
  b.p[0] = *(const v8bf*)(p);
  b.p[1] = *(const v8bf*)(p + 8);
  return b.v;
}
#define WMMA_BF16(A, B, C) \
  __builtin_amdgcn_wmma_f32_16x16x32_bf16(false, (A), false, (B), (short)0, (C), false, false)

// ---------------------------------------------------------------------------
// Kernel 3: gates GEMM.  [8192 x 512] bf16 @ [512 x 512] bf16 -> u|r logits.
// Block = 8 waves, 128(M) x 128(N) tile. Wave: 2 M-subtiles x 4 N-subtiles
// = 8 accumulators. B staged in LDS per 128-K chunk; A software-pipelined
// from global. Epilogue: sigmoid -> update_gate (f32) / bf16(reset*cs).
// ---------------------------------------------------------------------------
__global__ __launch_bounds__(256) void gemm_gates(
    const __bf16* __restrict__ acsb, const __bf16* __restrict__ wurt,
    const float* __restrict__ bu, const float* __restrict__ br,
    const float* __restrict__ cs,
    float* __restrict__ ug, __bf16* __restrict__ hin2) {
  __shared__ __bf16 ldsB[128 * LDS_STRIDE];   // ~34.8 KB
  const int tid  = threadIdx.x;
  const int lane = tid & 31;
  const int wave = tid >> 5;
  const int lm   = lane & 15;
  const int h    = lane >> 4;
  const int m0   = blockIdx.x * 128 + (wave >> 1) * 16;
  const int m1   = m0 + 64;
  const int nwl  = (wave & 1) * 64;          // n offset within tile
  const int nb   = blockIdx.y * 128;         // global n base of tile

  v8f acc[2][4] = {};
  const __bf16* ar0 = acsb + (m0 + lm) * KD;
  const __bf16* ar1 = acsb + (m1 + lm) * KD;

  for (int kc = 0; kc < KD; kc += 128) {
    __syncthreads();
#pragma unroll
    for (int i = 0; i < 8; ++i) {            // cooperative B-chunk fill: 32KB
      int cid = tid + i * 256;
      int col = cid >> 4;
      int off = (cid & 15) * 8;
      *(v8bf*)&ldsB[col * LDS_STRIDE + off] =
          *(const v8bf*)(wurt + (nb + col) * KD + kc + off);
    }
    __syncthreads();

    v16bf a0 = load_a_frag(ar0 + kc, h);
    v16bf a1 = load_a_frag(ar1 + kc, h);
#pragma unroll
    for (int s = 0; s < 4; ++s) {
      v16bf na0 = a0, na1 = a1;
      if (s < 3) {                            // prefetch next K-step's A frags
        na0 = load_a_frag(ar0 + kc + (s + 1) * 32, h);
        na1 = load_a_frag(ar1 + kc + (s + 1) * 32, h);
      }
      const int k0 = s * 32;
#pragma unroll
      for (int j = 0; j < 4; ++j) {
        v16bf b = load_b_lds(&ldsB[(nwl + j * 16 + lm) * LDS_STRIDE + k0 + 16 * h]);
        acc[0][j] = WMMA_BF16(a0, b, acc[0][j]);
        acc[1][j] = WMMA_BF16(a1, b, acc[1][j]);
      }
      a0 = na0; a1 = na1;
    }
  }

#pragma unroll
  for (int t = 0; t < 2; ++t) {
    int mbase = (t == 0) ? m0 : m1;
#pragma unroll
    for (int j = 0; j < 4; ++j) {
      int gn = nb + nwl + j * 16 + lm;
#pragma unroll
      for (int e = 0; e < 8; ++e) {
        int gm = mbase + e + 8 * h;
        float a = acc[t][j][e];
        if (gn < Dn) {
          ug[gm * Dn + gn] = sigmoidf_fast(a + bu[gn]);
        } else {
          int nr = gn - Dn;
          float r = sigmoidf_fast(a + br[nr]);
          hin2[gm * Dn + nr] = to_bf16(r * cs[gm * Dn + nr]);
        }
      }
    }
  }
}

// ---------------------------------------------------------------------------
// Kernel 4: hidden GEMM + fused final blend.
// A operand: k<256 -> acsb[:, :256] (activation), k>=256 -> hin2 (r*cs bf16).
// Epilogue: h = tanh(acc + bh); out = (1-u)*cs + u*h.
// ---------------------------------------------------------------------------
__global__ __launch_bounds__(256) void gemm_hidden(
    const __bf16* __restrict__ acsb, const __bf16* __restrict__ hin2,
    const __bf16* __restrict__ wht, const float* __restrict__ bh,
    const float* __restrict__ cs, const float* __restrict__ ug,
    float* __restrict__ out) {
  __shared__ __bf16 ldsB[128 * LDS_STRIDE];
  const int tid  = threadIdx.x;
  const int lane = tid & 31;
  const int wave = tid >> 5;
  const int lm   = lane & 15;
  const int h    = lane >> 4;
  const int m0   = blockIdx.x * 128 + (wave >> 1) * 16;
  const int m1   = m0 + 64;
  const int nwl  = (wave & 1) * 64;
  const int nb   = blockIdx.y * 128;

  v8f acc[2][4] = {};

  for (int kc = 0; kc < KD; kc += 128) {
    __syncthreads();
#pragma unroll
    for (int i = 0; i < 8; ++i) {
      int cid = tid + i * 256;
      int col = cid >> 4;
      int off = (cid & 15) * 8;
      *(v8bf*)&ldsB[col * LDS_STRIDE + off] =
          *(const v8bf*)(wht + (nb + col) * KD + kc + off);
    }
    __syncthreads();

    // A row base for this 128-K chunk (never straddles the 256 boundary)
    const __bf16* r0;
    const __bf16* r1;
    if (kc < Dn) {
      r0 = acsb + (m0 + lm) * KD + kc;
      r1 = acsb + (m1 + lm) * KD + kc;
    } else {
      r0 = hin2 + (m0 + lm) * Dn + (kc - Dn);
      r1 = hin2 + (m1 + lm) * Dn + (kc - Dn);
    }

    v16bf a0 = load_a_frag(r0, h);
    v16bf a1 = load_a_frag(r1, h);
#pragma unroll
    for (int s = 0; s < 4; ++s) {
      v16bf na0 = a0, na1 = a1;
      if (s < 3) {
        na0 = load_a_frag(r0 + (s + 1) * 32, h);
        na1 = load_a_frag(r1 + (s + 1) * 32, h);
      }
      const int k0 = s * 32;
#pragma unroll
      for (int j = 0; j < 4; ++j) {
        v16bf b = load_b_lds(&ldsB[(nwl + j * 16 + lm) * LDS_STRIDE + k0 + 16 * h]);
        acc[0][j] = WMMA_BF16(a0, b, acc[0][j]);
        acc[1][j] = WMMA_BF16(a1, b, acc[1][j]);
      }
      a0 = na0; a1 = na1;
    }
  }

#pragma unroll
  for (int t = 0; t < 2; ++t) {
    int mbase = (t == 0) ? m0 : m1;
#pragma unroll
    for (int j = 0; j < 4; ++j) {
      int gn = nb + nwl + j * 16 + lm;
#pragma unroll
      for (int e = 0; e < 8; ++e) {
        int gm  = mbase + e + 8 * h;
        float hv = tanhf(acc[t][j][e] + bh[gn]);
        float u  = ug[gm * Dn + gn];
        float cv = cs[gm * Dn + gn];
        out[gm * Dn + gn] = (1.0f - u) * cv + u * hv;
      }
    }
  }
}

// ---------------------------------------------------------------------------
extern "C" void kernel_launch(void* const* d_in, const int* in_sizes, int n_in,
                              void* d_out, int out_size, void* d_ws, size_t ws_size,
                              hipStream_t stream) {
  (void)in_sizes; (void)n_in; (void)out_size; (void)ws_size;
  const float* cs = (const float*)d_in[0];
  const float* em = (const float*)d_in[1];
  const int*   an = (const int*)d_in[2];
  const int*   ae = (const int*)d_in[3];
  const float* Wu = (const float*)d_in[4];
  const float* bu = (const float*)d_in[5];
  const float* Wr = (const float*)d_in[6];
  const float* br = (const float*)d_in[7];
  const float* Wh = (const float*)d_in[8];
  const float* bh = (const float*)d_in[9];
  float* out = (float*)d_out;

  // workspace carve-up (~22 MB total)
  char* p = (char*)d_ws;
  __bf16* acsb = (__bf16*)p; p += (size_t)Mtot * KD * 2;   // 8 MB  [act|cs] bf16
  __bf16* wurt = (__bf16*)p; p += (size_t)512 * 512 * 2;   // 512 KB Wu|Wr transposed
  __bf16* wht  = (__bf16*)p; p += (size_t)256 * 512 * 2;   // 256 KB Wh transposed
  float*  ug   = (float*)p;  p += (size_t)Mtot * Dn * 4;   // 8 MB  update gate f32
  __bf16* hin2 = (__bf16*)p; p += (size_t)Mtot * Dn * 2;   // 4 MB  r*cs bf16

  prep_weights<<<(512 * 512 + 256 * 512 + 255) / 256, 256, 0, stream>>>(
      Wu, Wr, Wh, wurt, wht);
  gather_sum<<<Mtot, 256, 0, stream>>>(cs, em, an, ae, acsb);

  dim3 blk(256);
  dim3 g_gates(Mtot / 128, (2 * Dn) / 128);  // 64 x 4
  dim3 g_hidden(Mtot / 128, Dn / 128);       // 64 x 2
  gemm_gates<<<g_gates, blk, 0, stream>>>(acsb, wurt, bu, br, cs, ug, hin2);
  gemm_hidden<<<g_hidden, blk, 0, stream>>>(acsb, hin2, wht, bh, cs, ug, out);
}
